// DecoupledDynamicRewireGNN_27702539059576
// MI455X (gfx1250) — matrix operand order repaired and checked by reading, not compiled, not run on hardware
//
#include <hip/hip_runtime.h>
#include <hip/hip_bf16.h>
#include <math.h>

#define NNODES 50000
#define NEDGES 600000
#define NCAND  600000
#define HDIM   128
#define NLAYER 3
#define NGRAPH 64
#define NCLASS 10

typedef __bf16 bf16_t;
typedef __attribute__((ext_vector_type(16))) bf16_t v16bf;
typedef __attribute__((ext_vector_type(8)))  float  v8f;

// ---------------------------------------------------------------------------
// Weight packing: W is [K][N] row-major fp32 -> Wt[N][K] bf16 (row-major K).
// Each lane's WMMA B-fragment becomes one contiguous 32-byte read.
// ---------------------------------------------------------------------------
__global__ void pack_wt_kernel(const float* __restrict__ W, bf16_t* __restrict__ out,
                               int K, int Nn) {
  int t = blockIdx.x * blockDim.x + threadIdx.x;
  if (t >= K * Nn) return;
  int n = t / K;
  int k = t - n * K;
  out[(long)n * K + k] = (bf16_t)W[(long)k * Nn + n];
}

// ---------------------------------------------------------------------------
// Row-gather WMMA GEMM:  out[M][128] = epilogue( gather(A)[M][KDIM] @ Wt + bias )
//   KDIM = 128: A row = in0[idx0[r]] (+ add0[r] elementwise)   (conv/int MLPs, e_emb)
//   KDIM = 256: A row = concat(in0[idx0[r]], in1[idx1[r]])     (add/del MLPs)
//   EPI 0: store z            EPI 1: store relu(z)
//   EPI 2: out[r] = sigmoid( relu(z) . w2 + w2b )   (fused H->1 second layer)
//
// Each wave owns TWO 16-row A stripes (32 rows x 128 cols): every B fragment
// loaded from the L2-resident packed weights feeds two back-to-back WMMAs,
// halving B-operand bandwidth per FLOP (the binding resource at WMMA rates).
// A/B use contiguous k = 16*(lane/16) + s in both (not the ISA table order); D follows CDNA5 ISA 7.12.2.
// ---------------------------------------------------------------------------
template<int KDIM, int EPI>
__global__ __launch_bounds__((KDIM == 256) ? 128 : 256)
void wmma_gemm_kernel(int M,
                      const float* __restrict__ in0, const int* __restrict__ idx0, int idx0_stride,
                      const float* __restrict__ add0,
                      const float* __restrict__ in1, const int* __restrict__ idx1, int idx1_stride,
                      const bf16_t* __restrict__ Wt,
                      const float* __restrict__ bias,
                      const float* __restrict__ w2, const float* __restrict__ w2b,
                      float* __restrict__ out) {
  const int WAVES   = (KDIM == 256) ? 4 : 8;
  const int MTILE   = WAVES * 32;        // 2 stripes of 16 rows per wave
  const int lane    = threadIdx.x & 31;
  const int wave    = threadIdx.x >> 5;
  const int mbase   = blockIdx.x * MTILE + wave * 32;
  const int halfsel = lane >> 4;         // 0: lanes 0-15, 1: lanes 16-31
  const int l16     = lane & 15;
  const int kbaseA  = halfsel * 16;      // A: K sub-offset per lane half
  const int kbaseB  = halfsel * 16;      // B: K sub-offset per lane half

  // Gather pointers for the two A stripes this lane supplies.
  const float* p0s[2];
  const float* a0s[2];
  const float* p1s[2];
#pragma unroll
  for (int s = 0; s < 2; ++s) {
    int rowA  = mbase + s * 16 + l16;
    int rowAc = (rowA < M) ? rowA : 0;   // clamp for safe gathers; stores guarded
    int g0 = idx0 ? idx0[(long)rowAc * idx0_stride] : rowAc;
    p0s[s] = in0 + (long)g0 * HDIM;
    a0s[s] = add0 ? (add0 + (long)rowAc * HDIM) : nullptr;
    p1s[s] = nullptr;
    if (KDIM == 256) {
      int g1 = idx1[(long)rowAc * idx1_stride];
      p1s[s] = in1 + (long)g1 * HDIM;
    }
  }

  v8f acc[2][8];
#pragma unroll
  for (int s = 0; s < 2; ++s)
#pragma unroll
    for (int t = 0; t < 8; ++t)
#pragma unroll
      for (int j = 0; j < 8; ++j) acc[s][t][j] = 0.0f;

#pragma unroll
  for (int kb = 0; kb < KDIM / 32; ++kb) {
    const int kg = kb * 32;
    // ---- build two A fragments (16x32 bf16) from gathered fp32 rows ----
    v16bf afrag[2];
#pragma unroll
    for (int s = 0; s < 2; ++s) {
      const float* srcp = p0s[s];
      const float* adda = a0s[s];
      int klocal = kg;
      if (KDIM == 256 && kg >= HDIM) { srcp = p1s[s]; adda = nullptr; klocal = kg - HDIM; }
      v8f lo = *(const v8f*)(srcp + klocal + kbaseA);        // K = kg+kbaseA .. +7
      v8f hi = *(const v8f*)(srcp + klocal + kbaseA + 8);    // K = kg+kbaseA+8 .. +7
      if (adda) {
        lo += *(const v8f*)(adda + klocal + kbaseA);
        hi += *(const v8f*)(adda + klocal + kbaseA + 8);
      }
#pragma unroll
      for (int j = 0; j < 8; ++j) {
        afrag[s][j]     = (bf16_t)lo[j];
        afrag[s][j + 8] = (bf16_t)hi[j];
      }
    }
    // prefetch next K-block of the weight stream (gfx1250 global_prefetch_b8)
    if (kb + 1 < KDIM / 32)
      __builtin_prefetch(Wt + (long)l16 * KDIM + kg + 32, 0, 1);
    // ---- 8 N-tiles of 16 columns; each B fragment feeds 2 WMMAs ----
#pragma unroll
    for (int nt = 0; nt < 8; ++nt) {
      const bf16_t* bp = Wt + (long)(nt * 16 + l16) * KDIM + kg + kbaseB;
      v16bf bfrag = *(const v16bf*)bp;   // 32 contiguous bytes -> B fragment
      acc[0][nt] = __builtin_amdgcn_wmma_f32_16x16x32_bf16(
          false, afrag[0], false, bfrag, (short)0, acc[0][nt], false, false);
      acc[1][nt] = __builtin_amdgcn_wmma_f32_16x16x32_bf16(
          false, afrag[1], false, bfrag, (short)0, acc[1][nt], false, false);
    }
  }

  if (EPI == 2) {
    // fused: relu(z) . w2  -> sigmoid
    float b2 = w2b[0];
#pragma unroll
    for (int s = 0; s < 2; ++s) {
      float psum[8];
#pragma unroll
      for (int r = 0; r < 8; ++r) psum[r] = 0.0f;
#pragma unroll
      for (int nt = 0; nt < 8; ++nt) {
        int col  = nt * 16 + l16;
        float b1 = bias[col];
        float wc = w2[col];
#pragma unroll
        for (int r = 0; r < 8; ++r) {
          float v = acc[s][nt][r] + b1;
          v = v > 0.0f ? v : 0.0f;
          psum[r] += v * wc;
        }
      }
#pragma unroll
      for (int r = 0; r < 8; ++r) {
        float v = psum[r];
        v += __shfl_xor(v, 1);
        v += __shfl_xor(v, 2);
        v += __shfl_xor(v, 4);
        v += __shfl_xor(v, 8);           // reduce within each 16-lane half
        if (l16 == 0) {
          int row = mbase + s * 16 + halfsel * 8 + r;   // D layout: M = halfsel*8 + r
          if (row < M) {
            float z = v + b2;
            out[row] = 1.0f / (1.0f + __expf(-z));
          }
        }
      }
    }
  } else {
#pragma unroll
    for (int s = 0; s < 2; ++s)
#pragma unroll
      for (int nt = 0; nt < 8; ++nt) {
        int col  = nt * 16 + l16;
        float b1 = bias[col];
#pragma unroll
        for (int r = 0; r < 8; ++r) {
          int row = mbase + s * 16 + halfsel * 8 + r;
          if (row < M) {
            float v = acc[s][nt][r] + b1;
            if (EPI == 1) v = v > 0.0f ? v : 0.0f;
            out[(long)row * HDIM + col] = v;
          }
        }
      }
  }
}

// ---------------------------------------------------------------------------
// Scatter message aggregation:
//   agg[didx[e]] += relu( xsrc[ridx[e]] + (eadd? eadd[e]:0) + (bvec? bvec:0) ) * wvec[e]
// One thread per (edge, 4 columns).
// ---------------------------------------------------------------------------
__global__ void msg_agg_kernel(int E,
                               const float* __restrict__ xsrc,
                               const int* __restrict__ ridx, int rstride,
                               const float* __restrict__ eadd,
                               const float* __restrict__ bvec,
                               const float* __restrict__ wvec,
                               const int* __restrict__ didx, int dstride,
                               float* __restrict__ agg) {
  long t = (long)blockIdx.x * blockDim.x + threadIdx.x;
  long total = (long)E * (HDIM / 4);
  if (t >= total) return;
  int e = (int)(t / (HDIM / 4));
  int c = (int)(t % (HDIM / 4)) * 4;
  int r = ridx[(long)e * rstride];
  int d = didx[(long)e * dstride];
  float w = wvec[e];
  const float* xp = xsrc + (long)r * HDIM + c;
  const float* ep = eadd ? (eadd + (long)e * HDIM + c) : nullptr;
  float* ap = agg + (long)d * HDIM + c;
#pragma unroll
  for (int u = 0; u < 4; ++u) {
    float v = xp[u];
    if (ep)   v += ep[u];
    if (bvec) v += bvec[c + u];
    v = fmaxf(v, 0.0f) * w;
    atomicAdd(ap + u, v);
  }
}

__global__ void zero_kernel(float* __restrict__ p, long n) {
  long t = (long)blockIdx.x * blockDim.x + threadIdx.x;
  if (t < n) p[t] = 0.0f;
}

__global__ void relu_add_kernel(float* __restrict__ x, const float* __restrict__ xn, long n) {
  long t = (long)blockIdx.x * blockDim.x + threadIdx.x;
  if (t < n) x[t] += fmaxf(xn[t], 0.0f);
}

__global__ void pool_kernel(const float* __restrict__ xd, const int* __restrict__ batch,
                            float* __restrict__ pooled, float* __restrict__ counts) {
  long t = (long)blockIdx.x * blockDim.x + threadIdx.x;
  long total = (long)NNODES * (HDIM / 4);
  if (t >= total) return;
  int nd = (int)(t / (HDIM / 4));
  int c  = (int)(t % (HDIM / 4)) * 4;
  int b  = batch[nd];
  const float* xp = xd + (long)nd * HDIM + c;
  float* pp = pooled + (long)b * HDIM + c;
#pragma unroll
  for (int u = 0; u < 4; ++u) atomicAdd(pp + u, xp[u]);
  if (c == 0) atomicAdd(counts + b, 1.0f);
}

__global__ void final_kernel(const float* __restrict__ pooled, const float* __restrict__ counts,
                             const float* __restrict__ Wf, const float* __restrict__ bfv,
                             float* __restrict__ out) {
  int t = blockIdx.x * blockDim.x + threadIdx.x;
  if (t >= NGRAPH * NCLASS) return;
  int g = t / NCLASS, j = t - g * NCLASS;
  float inv = 1.0f / fmaxf(counts[g], 1.0f);
  float s = 0.0f;
  for (int h = 0; h < HDIM; ++h) s += pooled[(long)g * HDIM + h] * Wf[(long)h * NCLASS + j];
  out[t] = s * inv + bfv[j];
}

// ---------------------------------------------------------------------------
extern "C" void kernel_launch(void* const* d_in, const int* in_sizes, int n_in,
                              void* d_out, int out_size, void* d_ws, size_t ws_size,
                              hipStream_t stream) {
  (void)in_sizes; (void)n_in; (void)out_size; (void)ws_size;

  const float* x           = (const float*)d_in[0];
  const float* edge_attr   = (const float*)d_in[1];
  const float* edge_weight = (const float*)d_in[2];
  const int*   edge_index  = (const int*)d_in[3];
  const int*   eci         = (const int*)d_in[4];
  const int*   batch       = (const int*)d_in[5];
  const float* We      = (const float*)d_in[6];
  const float* be      = (const float*)d_in[7];
  const float* conv_W1 = (const float*)d_in[8];
  const float* conv_b1 = (const float*)d_in[9];
  const float* conv_W2 = (const float*)d_in[10];
  const float* conv_b2 = (const float*)d_in[11];
  const float* add_A1  = (const float*)d_in[12];
  const float* add_a1  = (const float*)d_in[13];
  const float* add_A2  = (const float*)d_in[14];
  const float* add_a2  = (const float*)d_in[15];
  const float* del_A1  = (const float*)d_in[16];
  const float* del_a1  = (const float*)d_in[17];
  const float* del_A2  = (const float*)d_in[18];
  const float* del_a2  = (const float*)d_in[19];
  const float* int_W1  = (const float*)d_in[20];
  const float* int_b1  = (const float*)d_in[21];
  const float* int_W2  = (const float*)d_in[22];
  const float* int_b2  = (const float*)d_in[23];
  const float* Wf      = (const float*)d_in[24];
  const float* bfv     = (const float*)d_in[25];

  const int* src = edge_index;
  const int* dst = edge_index + NEDGES;
  const int* c0  = eci;      // stride 2
  const int* c1  = eci + 1;  // stride 2

  size_t off = 0;
  auto wsalloc = [&](size_t bytes) -> void* {
    off = (off + 255) & ~(size_t)255;
    void* p = (char*)d_ws + off;
    off += bytes;
    return p;
  };
  float* e_emb  = (float*)wsalloc((size_t)NEDGES * HDIM * 4);
  float* x_up   = (float*)wsalloc((size_t)NNODES * HDIM * 4);
  float* x_down = (float*)wsalloc((size_t)NNODES * HDIM * 4);
  float* x_new  = (float*)wsalloc((size_t)NNODES * HDIM * 4);
  float* hidden = (float*)wsalloc((size_t)NNODES * HDIM * 4);
  float* agg    = (float*)wsalloc((size_t)NNODES * HDIM * 4);
  float* w_add  = (float*)wsalloc((size_t)NCAND * 4);
  float* w_keep = (float*)wsalloc((size_t)NEDGES * 4);
  float* pooled = (float*)wsalloc((size_t)NGRAPH * HDIM * 4);
  float* counts = (float*)wsalloc((size_t)NGRAPH * 4);
  bf16_t* wt_we = (bf16_t*)wsalloc((size_t)HDIM * HDIM * 2);
  bf16_t* wt_c1 = (bf16_t*)wsalloc((size_t)NLAYER * HDIM * HDIM * 2);
  bf16_t* wt_c2 = (bf16_t*)wsalloc((size_t)NLAYER * HDIM * HDIM * 2);
  bf16_t* wt_i1 = (bf16_t*)wsalloc((size_t)NLAYER * HDIM * HDIM * 2);
  bf16_t* wt_i2 = (bf16_t*)wsalloc((size_t)NLAYER * HDIM * HDIM * 2);
  bf16_t* wt_a1 = (bf16_t*)wsalloc((size_t)NLAYER * 2 * HDIM * HDIM * 2);
  bf16_t* wt_d1 = (bf16_t*)wsalloc((size_t)NLAYER * 2 * HDIM * HDIM * 2);

  auto pack = [&](const float* w, bf16_t* o, int K, int Nn) {
    int tot = K * Nn;
    pack_wt_kernel<<<(tot + 255) / 256, 256, 0, stream>>>(w, o, K, Nn);
  };
  pack(We, wt_we, HDIM, HDIM);
  for (int i = 0; i < NLAYER; ++i) {
    pack(conv_W1 + (size_t)i * HDIM * HDIM, wt_c1 + (size_t)i * HDIM * HDIM, HDIM, HDIM);
    pack(conv_W2 + (size_t)i * HDIM * HDIM, wt_c2 + (size_t)i * HDIM * HDIM, HDIM, HDIM);
    pack(int_W1  + (size_t)i * HDIM * HDIM, wt_i1 + (size_t)i * HDIM * HDIM, HDIM, HDIM);
    pack(int_W2  + (size_t)i * HDIM * HDIM, wt_i2 + (size_t)i * HDIM * HDIM, HDIM, HDIM);
    pack(add_A1 + (size_t)i * 2 * HDIM * HDIM, wt_a1 + (size_t)i * 2 * HDIM * HDIM, 2 * HDIM, HDIM);
    pack(del_A1 + (size_t)i * 2 * HDIM * HDIM, wt_d1 + (size_t)i * 2 * HDIM * HDIM, 2 * HDIM, HDIM);
  }

  hipMemcpyAsync(x_up,   x, (size_t)NNODES * HDIM * 4, hipMemcpyDeviceToDevice, stream);
  hipMemcpyAsync(x_down, x, (size_t)NNODES * HDIM * 4, hipMemcpyDeviceToDevice, stream);

  const long NH = (long)NNODES * HDIM;
  const int ZB  = (int)((NH + 255) / 256);
  const int EG  = (int)(((long)NEDGES * (HDIM / 4) + 255) / 256);
  const int CG  = (int)(((long)NCAND  * (HDIM / 4) + 255) / 256);
  // workgroup M coverage: K=128 kernels: 8 waves * 32 rows = 256; K=256: 4 * 32 = 128
  const int GN  = (NNODES + 255) / 256;
  const int GE  = (NEDGES + 255) / 256;
  const int GE2 = (NEDGES + 127) / 128;
  const int GC2 = (NCAND  + 127) / 128;

  // e_emb = edge_attr @ We + be
  wmma_gemm_kernel<128, 0><<<GE, 256, 0, stream>>>(
      NEDGES, edge_attr, nullptr, 1, nullptr, nullptr, nullptr, 1,
      wt_we, be, nullptr, nullptr, e_emb);

  for (int i = 0; i < NLAYER; ++i) {
    // ---- up path ----
    zero_kernel<<<ZB, 256, 0, stream>>>(agg, NH);
    msg_agg_kernel<<<EG, 256, 0, stream>>>(NEDGES, x_up, src, 1, e_emb, nullptr,
                                           edge_weight, dst, 1, agg);
    wmma_gemm_kernel<128, 1><<<GN, 256, 0, stream>>>(
        NNODES, x_up, nullptr, 1, agg, nullptr, nullptr, 1,
        wt_c1 + (size_t)i * HDIM * HDIM, conv_b1 + (size_t)i * HDIM,
        nullptr, nullptr, hidden);
    wmma_gemm_kernel<128, 0><<<GN, 256, 0, stream>>>(
        NNODES, hidden, nullptr, 1, nullptr, nullptr, nullptr, 1,
        wt_c2 + (size_t)i * HDIM * HDIM, conv_b2 + (size_t)i * HDIM,
        nullptr, nullptr, x_new);
    // w_add = sigmoid(mlp2(concat(x_new[c0], x_new[c1])))
    wmma_gemm_kernel<256, 2><<<GC2, 128, 0, stream>>>(
        NCAND, x_new, c0, 2, nullptr, x_new, c1, 2,
        wt_a1 + (size_t)i * 2 * HDIM * HDIM, add_a1 + (size_t)i * HDIM,
        add_A2 + (size_t)i * HDIM, add_a2 + i, w_add);
    // w_keep = sigmoid(mlp2(concat(x_new[src], x_new[dst])))
    wmma_gemm_kernel<256, 2><<<GE2, 128, 0, stream>>>(
        NEDGES, x_new, src, 1, nullptr, x_new, dst, 1,
        wt_d1 + (size_t)i * 2 * HDIM * HDIM, del_a1 + (size_t)i * HDIM,
        del_A2 + (size_t)i * HDIM, del_a2 + i, w_keep);
    relu_add_kernel<<<ZB, 256, 0, stream>>>(x_up, x_new, NH);

    // ---- down path ----
    zero_kernel<<<ZB, 256, 0, stream>>>(agg, NH);
    msg_agg_kernel<<<EG, 256, 0, stream>>>(NEDGES, x_down, src, 1, e_emb, nullptr,
                                           w_keep, dst, 1, agg);
    msg_agg_kernel<<<CG, 256, 0, stream>>>(NCAND, x_down, c0, 2, nullptr, be,
                                           w_add, c1, 2, agg);
    wmma_gemm_kernel<128, 1><<<GN, 256, 0, stream>>>(
        NNODES, x_down, nullptr, 1, agg, nullptr, nullptr, 1,
        wt_i1 + (size_t)i * HDIM * HDIM, int_b1 + (size_t)i * HDIM,
        nullptr, nullptr, hidden);
    wmma_gemm_kernel<128, 0><<<GN, 256, 0, stream>>>(
        NNODES, hidden, nullptr, 1, nullptr, nullptr, nullptr, 1,
        wt_i2 + (size_t)i * HDIM * HDIM, int_b2 + (size_t)i * HDIM,
        nullptr, nullptr, x_new);
    relu_add_kernel<<<ZB, 256, 0, stream>>>(x_down, x_new, NH);
  }

  // ---- pooling + head ----
  zero_kernel<<<(NGRAPH * HDIM + 255) / 256, 256, 0, stream>>>(pooled, (long)NGRAPH * HDIM);
  zero_kernel<<<1, 256, 0, stream>>>(counts, (long)NGRAPH);
  pool_kernel<<<(int)(((long)NNODES * (HDIM / 4) + 255) / 256), 256, 0, stream>>>(
      x_down, batch, pooled, counts);
  final_kernel<<<(NGRAPH * NCLASS + 255) / 256, 256, 0, stream>>>(
      pooled, counts, Wf, bfv, (float*)d_out);
}
